// LSTM_DNC_59468117180814
// MI455X (gfx1250) — matrix operand order repaired and checked by reading, not compiled, not run on hardware
//
#include <hip/hip_runtime.h>
#include <cstdint>
#include <cstddef>

// ---- problem constants ----
#define Hd    512
#define Td    256
#define Bd    64
#define Rr    4
#define Wd    32
#define Nn    64
#define HIDC  759          // H + IFACE
#define HIDCP 768          // padded controller hidden
#define GC    3036         // 4*HID_C
#define GCP   3040         // padded to /16
#define GE    2048         // 4*H
#define KE    1024         // encoder concat K = H + H
#define KC    1408         // controller concat K = 512+128+759 -> pad 1408
#define INC   640          // H + R*W
#define EPSF  1e-6f

typedef __attribute__((ext_vector_type(16))) __bf16      v16bf;
typedef __attribute__((ext_vector_type(8)))  float        v8f;
typedef __attribute__((ext_vector_type(4)))  unsigned int u32x4;

union FragU { v16bf v; u32x4 q[2]; unsigned int u[8]; };

__device__ __forceinline__ unsigned short f2bf(float f) {
  union { float f; unsigned int u; } v; v.f = f;
  unsigned int u = v.u;
  unsigned int r = u + 0x7FFFu + ((u >> 16) & 1u);   // round-to-nearest-even
  return (unsigned short)(r >> 16);
}
__device__ __forceinline__ float sigmf(float x) { return 1.f / (1.f + expf(-x)); }
__device__ __forceinline__ float softplusf(float x) { return (x > 20.f) ? x : log1pf(expf(x)); }

// ================= weight / bias prep =================
__global__ void prep_wcat_e(const float* __restrict__ wih, const float* __restrict__ whh,
                            unsigned short* __restrict__ wc) {
  int idx = blockIdx.x * blockDim.x + threadIdx.x;
  if (idx >= GE * KE) return;
  int g = idx / KE, k = idx % KE;
  float v = (k < Hd) ? wih[g * Hd + k] : whh[g * Hd + (k - Hd)];
  wc[idx] = f2bf(v);
}

__global__ void prep_wcat_c(const float* __restrict__ wih, const float* __restrict__ whh,
                            unsigned short* __restrict__ wc) {
  int idx = blockIdx.x * blockDim.x + threadIdx.x;
  if (idx >= GCP * KC) return;
  int g = idx / KC, k = idx % KC;
  float v = 0.f;
  if (g < GC) {
    if (k < INC)              v = wih[g * INC + k];           // [e | rv] part
    else if (k < INC + HIDC)  v = whh[g * HIDC + (k - INC)];  // h part
  }
  wc[idx] = f2bf(v);
}

__global__ void prep_bias_c(const float* __restrict__ b, float* __restrict__ bp) {
  int idx = blockIdx.x * blockDim.x + threadIdx.x;
  if (idx >= GCP) return;
  bp[idx] = (idx < GC) ? b[idx] : 0.f;
}

__global__ void zero_f32(float* __restrict__ p, int n) {
  int idx = blockIdx.x * blockDim.x + threadIdx.x;
  if (idx < n) p[idx] = 0.f;
}

// ================= WMMA bf16 GEMM: C(64 x Ng) = A(64 x K) * W(Ng x K)^T + bias =================
// One wave per 16x16 output tile; K stepped by 32 using v_wmma_f32_16x16x32_bf16.
// Fragment layouts follow CDNA5 ISA 7.12.2 (16-bit A 16x32, B 32x16, 32-bit C 16x16).
__global__ __launch_bounds__(32) void gemm_bf16_wmma(
    const unsigned short* __restrict__ A, int lda,    // 64 x K row-major bf16
    const unsigned short* __restrict__ Wt, int ldw,   // Ng x K row-major bf16 (acts as B^T)
    const float* __restrict__ bias,
    float* __restrict__ Cout, int ldc, int K) {
  const int l      = threadIdx.x;     // 0..31
  const int half   = l >> 4;          // 0/1
  const int lane16 = l & 15;
  const int m0 = blockIdx.y * 16;
  const int n0 = blockIdx.x * 16;

  const unsigned short* Arow = A  + (size_t)(m0 + lane16) * lda;
  const unsigned short* Wrow = Wt + (size_t)(n0 + lane16) * ldw;

  v8f acc = {};
  for (int k0 = 0; k0 < K; k0 += 32) {
    __builtin_prefetch(Wrow + k0 + 128, 0, 1);
    __builtin_prefetch(Arow + k0 + 128, 0, 1);
    FragU a, b;
    // A 16x32 bf16: VGPR p holds K pair {16*(p/4) + 8*half + 2*(p%4), +1}
#pragma unroll
    for (int p = 0; p < 8; ++p) {
      int kk = k0 + 16 * (p >> 2) + 8 * half + 2 * (p & 3);
      a.u[p] = *(const unsigned int*)(Arow + kk);
    }
    // B 32x16 bf16: lane holds column n=lane16, K = k0 + 8*half .. +7 and k0 + 16 + 8*half .. +7
    int kb = k0 + 8 * half;
    b.q[0] = *(const u32x4*)(Wrow + kb);
    b.q[1] = *(const u32x4*)(Wrow + kb + 16);
    acc = __builtin_amdgcn_wmma_f32_16x16x32_bf16(false, a.v, false, b.v,
                                                  (short)0, acc, false, false);
  }
  // C: VGPR r -> row r + 8*half, col lane16
#pragma unroll
  for (int r = 0; r < 8; ++r) {
    int m = m0 + r + 8 * half;
    int n = n0 + lane16;
    Cout[(size_t)m * ldc + n] = acc[r] + bias[n];
  }
}

// ================= pack concatenated GEMM inputs (fp32 -> bf16) =================
__global__ void pack_enc(const float* __restrict__ x, const float* __restrict__ h,
                         unsigned short* __restrict__ xc, int t) {
  int idx = blockIdx.x * blockDim.x + threadIdx.x;
  if (idx >= Bd * KE) return;
  int b = idx >> 10, k = idx & 1023;
  float v = (k < Hd) ? x[((size_t)b * Td + t) * Hd + k] : h[b * Hd + (k - Hd)];
  xc[idx] = f2bf(v);
}

__global__ void pack_ctl(const unsigned short* __restrict__ enc, const float* __restrict__ rv,
                         const float* __restrict__ h, unsigned short* __restrict__ xc, int t) {
  int idx = blockIdx.x * blockDim.x + threadIdx.x;
  if (idx >= Bd * KC) return;
  int b = idx / KC, k = idx % KC;
  unsigned short v;
  if (k < Hd)                 v = enc[(size_t)t * Bd * Hd + b * Hd + k];
  else if (k < INC)           v = f2bf(rv[b * (Rr * Wd) + (k - Hd)]);
  else if (k < INC + HIDC)    v = f2bf(h[b * HIDCP + (k - INC)]);
  else                        v = 0;
  xc[idx] = v;
}

// ================= LSTM pointwise =================
__global__ void lstm_point_enc(const float* __restrict__ g, float* __restrict__ h,
                               float* __restrict__ c, unsigned short* __restrict__ enc, int t) {
  int idx = blockIdx.x * blockDim.x + threadIdx.x;
  if (idx >= Bd * Hd) return;
  int b = idx >> 9, j = idx & 511;
  const float* gb = g + (size_t)b * GE;
  float gi = gb[j], gf = gb[Hd + j], gg = gb[2 * Hd + j], go = gb[3 * Hd + j];
  float cn = sigmf(gf) * c[idx] + sigmf(gi) * tanhf(gg);
  float hn = sigmf(go) * tanhf(cn);
  c[idx] = cn; h[idx] = hn;
  enc[(size_t)t * Bd * Hd + idx] = f2bf(hn);
}

__global__ void lstm_point_ctl(const float* __restrict__ g, float* __restrict__ h,
                               float* __restrict__ c, float* __restrict__ out, int t) {
  int idx = blockIdx.x * blockDim.x + threadIdx.x;
  if (idx >= Bd * HIDCP) return;
  int b = idx / HIDCP, j = idx % HIDCP;
  if (j >= HIDC) return;
  const float* gb = g + (size_t)b * GCP;
  float gi = gb[j], gf = gb[HIDC + j], gg = gb[2 * HIDC + j], go = gb[3 * HIDC + j];
  float cn = sigmf(gf) * c[idx] + sigmf(gi) * tanhf(gg);
  float hn = sigmf(go) * tanhf(cn);
  c[idx] = cn; h[idx] = hn;
  if (j < Hd) out[((size_t)b * Td + t) * Hd + j] = hn;
}

// ================= DNC memory module: one 64-thread block per batch element =================
__global__ __launch_bounds__(64) void dnc_memory_kernel(
    const float* __restrict__ h_c,
    float* __restrict__ Mg, float* __restrict__ ug, float* __restrict__ pg,
    float* __restrict__ Lg, float* __restrict__ wrg, float* __restrict__ wwg,
    float* __restrict__ rvg) {
  const int b = blockIdx.x, tid = threadIdx.x, n = tid;
  __shared__ float rk[Rr][Wd], rs[Rr], wkey[Wd], er[Wd], wv[Wd], fg[Rr], rm[Rr][3];
  __shared__ float sAg, sWg, sWstr;
  __shared__ float Ml[Nn][Wd];          // 8 KB
  __shared__ float Ll[Nn][Nn];          // 16 KB
  __shared__ float uS[Nn], suS[Nn], cpS[Nn], wwNew[Nn], wwOld[Nn], pS[Nn];
  __shared__ float wrO[Rr][Nn], wrN[Rr][Nn];
  __shared__ float red[Nn];

  // ---- parse interface vector xi = h[:, 512:] ----
  const float* xi = h_c + (size_t)b * HIDCP + Hd;
  for (int idx = tid; idx < 247; idx += 64) {
    float v = xi[idx];
    if (idx < 128)       rk[idx >> 5][idx & 31] = tanhf(v);
    else if (idx < 132)  rs[idx - 128] = softplusf(v);
    else if (idx < 164)  wkey[idx - 132] = tanhf(v);
    else if (idx == 164) sWstr = softplusf(v);
    else if (idx < 197)  er[idx - 165] = sigmf(v);
    else if (idx < 229)  wv[idx - 197] = tanhf(v);
    else if (idx < 233)  fg[idx - 229] = sigmf(v);
    else if (idx == 233) sAg = sigmf(v);
    else if (idx == 234) sWg = sigmf(v);
    else                 rm[(idx - 235) / 3][(idx - 235) % 3] = v;  // raw
  }
  // ---- load state ----
  for (int w = 0; w < Wd; ++w) Ml[n][w] = Mg[((size_t)b * Nn + n) * Wd + w];
  for (int r = 0; r < Rr; ++r) wrO[r][n] = wrg[((size_t)b * Rr + r) * Nn + n];
  for (int m = 0; m < Nn; ++m) Ll[n][m] = Lg[((size_t)b * Nn + n) * Nn + m];
  uS[n] = ug[b * Nn + n]; pS[n] = pg[b * Nn + n]; wwOld[n] = wwg[b * Nn + n];
  __syncthreads();

  if (tid < Rr) {  // read-mode softmax (3-way)
    float a0 = rm[tid][0], a1 = rm[tid][1], a2 = rm[tid][2];
    float mx = fmaxf(a0, fmaxf(a1, a2));
    float e0 = expf(a0 - mx), e1 = expf(a1 - mx), e2 = expf(a2 - mx);
    float s = e0 + e1 + e2;
    rm[tid][0] = e0 / s; rm[tid][1] = e1 / s; rm[tid][2] = e2 / s;
  }
  // ---- retention + usage ----
  float psi = 1.f;
  for (int r = 0; r < Rr; ++r) psi *= (1.f - fg[r] * wrO[r][n]);
  float un = (uS[n] + wwOld[n] - uS[n] * wwOld[n]) * psi;
  uS[n] = un;
  __syncthreads();
  // ---- stable argsort via rank (O(N^2)) ----
  int rank = 0;
  for (int k = 0; k < Nn; ++k) {
    float uk = uS[k];
    rank += (uk < un) || (uk == un && k < n);
  }
  suS[rank] = un;
  __syncthreads();
  if (tid == 0) { float c = 1.f; for (int j = 0; j < Nn; ++j) { cpS[j] = c; c *= suS[j]; } }
  __syncthreads();
  float aAlloc = (1.f - un) * cpS[rank];
  // ---- write content weights on OLD M ----
  float mn2 = 0.f; for (int w = 0; w < Wd; ++w) mn2 += Ml[n][w] * Ml[n][w];
  float rM = rsqrtf(mn2 + EPSF);
  float k2 = 0.f; for (int w = 0; w < Wd; ++w) k2 += wkey[w] * wkey[w];
  float rK = rsqrtf(k2 + EPSF);
  float dot = 0.f; for (int w = 0; w < Wd; ++w) dot += wkey[w] * Ml[n][w];
  float sim = dot * rM * rK * sWstr;
  red[n] = sim; __syncthreads();
  for (int s = 32; s > 0; s >>= 1) { if (tid < s) red[tid] = fmaxf(red[tid], red[tid + s]); __syncthreads(); }
  float mx = red[0]; __syncthreads();
  float ex = expf(sim - mx);
  red[n] = ex; __syncthreads();
  for (int s = 32; s > 0; s >>= 1) { if (tid < s) red[tid] += red[tid + s]; __syncthreads(); }
  float cw = ex / red[0]; __syncthreads();
  // ---- write weighting, M update, L update, p update ----
  float wwn = sWg * (sAg * aAlloc + (1.f - sAg) * cw);
  wwNew[n] = wwn;
  __syncthreads();
  for (int w = 0; w < Wd; ++w)
    Ml[n][w] = Ml[n][w] * (1.f - wwn * er[w]) + wwn * wv[w];
  for (int j = 0; j < Nn; ++j) {
    float lv = (1.f - wwn - wwNew[j]) * Ll[n][j] + wwn * pS[j];
    Ll[n][j] = (j == n) ? 0.f : lv;
  }
  red[n] = wwn; __syncthreads();
  for (int s = 32; s > 0; s >>= 1) { if (tid < s) red[tid] += red[tid + s]; __syncthreads(); }
  float sWW = red[0]; __syncthreads();
  pS[n] = (1.f - sWW) * pS[n] + wwn;
  __syncthreads();
  // ---- read content weights on NEW M, forward/backward via NEW L, OLD wr ----
  float mn2b = 0.f; for (int w = 0; w < Wd; ++w) mn2b += Ml[n][w] * Ml[n][w];
  float rM2 = rsqrtf(mn2b + EPSF);
  for (int r = 0; r < Rr; ++r) {
    float kk2 = 0.f; for (int w = 0; w < Wd; ++w) kk2 += rk[r][w] * rk[r][w];
    float rKr = rsqrtf(kk2 + EPSF);
    float d = 0.f; for (int w = 0; w < Wd; ++w) d += rk[r][w] * Ml[n][w];
    float simr = d * rM2 * rKr * rs[r];
    red[n] = simr; __syncthreads();
    for (int s = 32; s > 0; s >>= 1) { if (tid < s) red[tid] = fmaxf(red[tid], red[tid + s]); __syncthreads(); }
    float mxr = red[0]; __syncthreads();
    float exr = expf(simr - mxr);
    red[n] = exr; __syncthreads();
    for (int s = 32; s > 0; s >>= 1) { if (tid < s) red[tid] += red[tid + s]; __syncthreads(); }
    float cr_ = exr / red[0]; __syncthreads();
    float fw = 0.f, bw = 0.f;
    for (int m = 0; m < Nn; ++m) { fw += Ll[n][m] * wrO[r][m]; bw += Ll[m][n] * wrO[r][m]; }
    wrN[r][n] = rm[r][0] * bw + rm[r][1] * cr_ + rm[r][2] * fw;
  }
  __syncthreads();
  // ---- read vectors rv = wr_new @ M_new ----
  for (int idx = tid; idx < Rr * Wd; idx += 64) {
    int r = idx >> 5, w = idx & 31;
    float s = 0.f;
    for (int m = 0; m < Nn; ++m) s += wrN[r][m] * Ml[m][w];
    rvg[b * (Rr * Wd) + idx] = s;
  }
  // ---- write back state ----
  for (int w = 0; w < Wd; ++w) Mg[((size_t)b * Nn + n) * Wd + w] = Ml[n][w];
  for (int m = 0; m < Nn; ++m) Lg[((size_t)b * Nn + n) * Nn + m] = Ll[n][m];
  for (int r = 0; r < Rr; ++r) wrg[((size_t)b * Rr + r) * Nn + n] = wrN[r][n];
  ug[b * Nn + n] = uS[n]; pg[b * Nn + n] = pS[n]; wwg[b * Nn + n] = wwNew[n];
}

// ================= host =================
extern "C" void kernel_launch(void* const* d_in, const int* in_sizes, int n_in,
                              void* d_out, int out_size, void* d_ws, size_t ws_size,
                              hipStream_t stream) {
  const float* x     = (const float*)d_in[0];
  const float* Wih_e = (const float*)d_in[1];
  const float* Whh_e = (const float*)d_in[2];
  const float* b_e   = (const float*)d_in[3];
  const float* Wih_c = (const float*)d_in[4];
  const float* Whh_c = (const float*)d_in[5];
  const float* b_c   = (const float*)d_in[6];
  float* out = (float*)d_out;

  char* ws = (char*)d_ws;
  size_t off = 0;
  auto take = [&](size_t bytes) -> char* {
    char* p = ws + off;
    off = (off + bytes + 255) & ~(size_t)255;
    return p;
  };
  unsigned short* WcatE = (unsigned short*)take((size_t)GE * KE * 2);
  unsigned short* WcatC = (unsigned short*)take((size_t)GCP * KC * 2);
  float*          biasC = (float*)take((size_t)GCP * 4);
  unsigned short* xcatE = (unsigned short*)take((size_t)Bd * KE * 2);
  unsigned short* xcatC = (unsigned short*)take((size_t)Bd * KC * 2);
  float*          gatesE = (float*)take((size_t)Bd * GE * 4);
  float*          gatesC = (float*)take((size_t)Bd * GCP * 4);
  unsigned short* enc = (unsigned short*)take((size_t)Td * Bd * Hd * 2);
  char* state0 = ws + off;
  float* h_e = (float*)take((size_t)Bd * Hd * 4);
  float* c_e = (float*)take((size_t)Bd * Hd * 4);
  float* h_c = (float*)take((size_t)Bd * HIDCP * 4);
  float* c_c = (float*)take((size_t)Bd * HIDCP * 4);
  float* Mm  = (float*)take((size_t)Bd * Nn * Wd * 4);
  float* uu  = (float*)take((size_t)Bd * Nn * 4);
  float* pp  = (float*)take((size_t)Bd * Nn * 4);
  float* LL  = (float*)take((size_t)Bd * Nn * Nn * 4);
  float* wr  = (float*)take((size_t)Bd * Rr * Nn * 4);
  float* wwv = (float*)take((size_t)Bd * Nn * 4);
  float* rv  = (float*)take((size_t)Bd * Rr * Wd * 4);
  int stateFloats = (int)(((ws + off) - state0) / 4);

  // one-time prep (deterministic, re-done every call)
  { int nn = GE * KE;  prep_wcat_e<<<(nn + 255) / 256, 256, 0, stream>>>(Wih_e, Whh_e, WcatE); }
  { int nn = GCP * KC; prep_wcat_c<<<(nn + 255) / 256, 256, 0, stream>>>(Wih_c, Whh_c, WcatC); }
  prep_bias_c<<<(GCP + 255) / 256, 256, 0, stream>>>(b_c, biasC);
  zero_f32<<<(stateFloats + 255) / 256, 256, 0, stream>>>((float*)state0, stateFloats);

  dim3 gE(GE / 16, Bd / 16);   // 128 x 4 tiles
  dim3 gC(GCP / 16, Bd / 16);  // 190 x 4 tiles

  // encoder scan
  for (int t = 0; t < Td; ++t) {
    pack_enc<<<(Bd * KE + 255) / 256, 256, 0, stream>>>(x, h_e, xcatE, t);
    gemm_bf16_wmma<<<gE, 32, 0, stream>>>(xcatE, KE, WcatE, KE, b_e, gatesE, GE, KE);
    lstm_point_enc<<<(Bd * Hd + 255) / 256, 256, 0, stream>>>(gatesE, h_e, c_e, enc, t);
  }
  // DNC scan
  for (int t = 0; t < Td; ++t) {
    pack_ctl<<<(Bd * KC + 255) / 256, 256, 0, stream>>>(enc, rv, h_c, xcatC, t);
    gemm_bf16_wmma<<<gC, 32, 0, stream>>>(xcatC, KC, WcatC, KC, biasC, gatesC, GCP, KC);
    lstm_point_ctl<<<(Bd * HIDCP + 255) / 256, 256, 0, stream>>>(gatesC, h_c, c_c, out, t);
    dnc_memory_kernel<<<Bd, 64, 0, stream>>>(h_c, Mm, uu, pp, LL, wr, wwv, rv);
  }
}
